// CollectiveBehaviorAnalyzer_4329327035075
// MI455X (gfx1250) — matrix operand order repaired and checked by reading, not compiled, not run on hardware
//
#include <hip/hip_runtime.h>
#include <hip/hip_bf16.h>

// ---------------------------------------------------------------------------
// CollectiveBehaviorAnalyzer for MI455X (gfx1250, wave32, WMMA)
// T=4096, N=1024, H=128 hardcoded to match setup_inputs().
// ---------------------------------------------------------------------------

#define T_DIM 4096
#define N_DIM 1024
#define NGRP  32                 // 32x32-of-16 tile groups per dimension
#define NGROUPS (NGRP * (NGRP + 1) / 2)   // 528 upper-triangular 32x32 groups
#define PI_BLOCKS 32
#define PI_ITERS  60

typedef __attribute__((ext_vector_type(16))) _Float16     v16h;
typedef __attribute__((ext_vector_type(8)))  _Float16     v8h;
typedef __attribute__((ext_vector_type(8)))  float        v8f;
typedef __attribute__((ext_vector_type(8)))  unsigned int v8u;

__device__ __forceinline__ v16h habs16(v16h x) {
    union { v16h h; v8u u; } t;
    t.h = x;
    t.u = t.u & 0x7FFF7FFFu;   // clear packed f16 sign bits
    return t.h;
}
__device__ __forceinline__ float clampf(float v, float lo, float hi) {
    return fminf(hi, fmaxf(lo, v));
}

// ---------------------------------------------------------------------------
// X [T,N] -> XT [N,T] (f32), 32x32 LDS tiles, fully coalesced both sides.
// ---------------------------------------------------------------------------
__global__ void __launch_bounds__(256) transpose_kernel(
    const float* __restrict__ X, float* __restrict__ XT)
{
    __shared__ float tile[32][33];
    const int n0 = blockIdx.x * 32, t0 = blockIdx.y * 32;
    const int tx = threadIdx.x, ty = threadIdx.y;
    #pragma unroll
    for (int it = 0; it < 4; ++it)
        tile[ty + it * 8][tx] = X[(size_t)(t0 + ty + it * 8) * N_DIM + n0 + tx];
    __syncthreads();
    #pragma unroll
    for (int it = 0; it < 4; ++it)
        XT[(size_t)(n0 + ty + it * 8) * T_DIM + t0 + tx] = tile[tx][ty + it * 8];
}

// ---------------------------------------------------------------------------
// Per-column stats from XT (contiguous rows): mean, 1/||x-mu||, nonzero count.
// ---------------------------------------------------------------------------
__global__ void __launch_bounds__(256) col_stats_kernel(
    const float* __restrict__ XT, float* __restrict__ colmean,
    float* __restrict__ colinv, float* __restrict__ nz)
{
    const int c = blockIdx.x;
    const int tid = threadIdx.x;
    const float* xp = XT + (size_t)c * T_DIM;
    float sum = 0.f, ssq = 0.f, n = 0.f;
    for (int t = tid * 4; t < T_DIM; t += 1024) {
        float4 v = *(const float4*)(xp + t);
        sum += v.x + v.y + v.z + v.w;
        ssq += v.x * v.x + v.y * v.y + v.z * v.z + v.w * v.w;
        n += ((v.x != 0.f) ? 1.f : 0.f) + ((v.y != 0.f) ? 1.f : 0.f)
           + ((v.z != 0.f) ? 1.f : 0.f) + ((v.w != 0.f) ? 1.f : 0.f);
    }
    __shared__ float r0[256], r1[256], r2[256];
    r0[tid] = sum; r1[tid] = ssq; r2[tid] = n;
    __syncthreads();
    for (int off = 128; off; off >>= 1) {
        if (tid < off) { r0[tid] += r0[tid + off]; r1[tid] += r1[tid + off]; r2[tid] += r2[tid + off]; }
        __syncthreads();
    }
    if (tid == 0) {
        float S = r0[0], SS = r1[0];
        float mu = S / (float)T_DIM;
        colmean[c] = mu;
        colinv[c]  = rsqrtf(fmaxf(SS - S * mu, 1e-30f));
        nz[c]      = r2[0];
    }
}

// ---------------------------------------------------------------------------
// Encode from XT: ZT = (x-mu)*inv (f16), ST = sign(x) (f16). 8 elems/thread.
// ---------------------------------------------------------------------------
__global__ void __launch_bounds__(256) encode_kernel(
    const float* __restrict__ XT, const float* __restrict__ colmean,
    const float* __restrict__ colinv,
    _Float16* __restrict__ ZT, _Float16* __restrict__ ST)
{
    const size_t idx = ((size_t)blockIdx.x * 256 + threadIdx.x) * 8;
    const int n = (int)(idx >> 12);          // idx / T_DIM
    const float mu = colmean[n], inv = colinv[n];
    float4 v0 = *(const float4*)(XT + idx);
    float4 v1 = *(const float4*)(XT + idx + 4);
    float xv[8] = { v0.x, v0.y, v0.z, v0.w, v1.x, v1.y, v1.z, v1.w };
    v8h z, s;
    #pragma unroll
    for (int i = 0; i < 8; ++i) {
        z[i] = (_Float16)((xv[i] - mu) * inv);
        s[i] = (_Float16)((xv[i] > 0.f) ? 1.f : ((xv[i] < 0.f) ? -1.f : 0.f));
    }
    *(v8h*)(ZT + idx) = z;
    *(v8h*)(ST + idx) = s;
}

// ---------------------------------------------------------------------------
// Cross-sectional std per row (unbiased). One wave per row, 8 waves/block.
// ---------------------------------------------------------------------------
__global__ void __launch_bounds__(256) row_std_kernel(
    const float* __restrict__ X, float* __restrict__ cs_std)
{
    const int wave = threadIdx.x >> 5, lane = threadIdx.x & 31;
    const int r = blockIdx.x * 8 + wave;
    const float* xp = X + (size_t)r * N_DIM;
    float sum = 0.f, ssq = 0.f;
    for (int j = lane * 4; j < N_DIM; j += 128) {
        float4 v = *(const float4*)(xp + j);
        sum += v.x + v.y + v.z + v.w;
        ssq += v.x * v.x + v.y * v.y + v.z * v.z + v.w * v.w;
    }
    for (int off = 16; off; off >>= 1) {
        sum += __shfl_down(sum, off);
        ssq += __shfl_down(ssq, off);
    }
    if (lane == 0) {
        float mu = sum / (float)N_DIM;
        cs_std[r] = sqrtf(fmaxf((ssq - sum * mu) / (float)(N_DIM - 1), 0.f));
    }
}

// ---------------------------------------------------------------------------
// WMMA fragment loaders from row-major [N, T] f16 (ISA 7.12.2 layouts).
// ---------------------------------------------------------------------------
__device__ __forceinline__ v16h load_fragA(const _Float16* rowp, int k, int hi) {
    v8h lo  = *(const v8h*)(rowp + k + 8 * hi);
    v8h hiv = *(const v8h*)(rowp + k + 16 + 8 * hi);
    return __builtin_shufflevector(lo, hiv, 0,1,2,3,4,5,6,7,8,9,10,11,12,13,14,15);
}
__device__ __forceinline__ v16h load_fragB(const _Float16* rowp, int k, int hi) {
    v8h lo  = *(const v8h*)(rowp + k + 8 * hi);
    v8h hiv = *(const v8h*)(rowp + k + 16 + 8 * hi);
    return __builtin_shufflevector(lo, hiv, 0,1,2,3,4,5,6,7,8,9,10,11,12,13,14,15);
}
__device__ __forceinline__ v8f wmma_f16(v16h a, v16h b, v8f c) {
    return __builtin_amdgcn_wmma_f32_16x16x32_f16(false, a, false, b, (short)0, c, false, false);
}
// map linear q in [0, NGROUPS) -> upper-triangular (gi, gj), gi <= gj
__device__ __forceinline__ void group_ij(int q, int& gi, int& gj) {
    int i = 0, rem = q;
    #pragma unroll 1
    while (rem >= (NGRP - i)) { rem -= (NGRP - i); ++i; }
    gi = i; gj = i + rem;
}

__device__ __forceinline__ void store_tile(
    float* __restrict__ M, v8f acc, int r0, int c0, int lane, bool mirror,
    float& s, float& sa)
{
    const int m = lane & 15, hi = lane >> 4;
    const int col = c0 + m;
    #pragma unroll
    for (int r = 0; r < 8; ++r) {
        int row = r0 + r + 8 * hi;
        float v = acc[r];
        M[(size_t)row * N_DIM + col] = v;
        if (mirror) M[(size_t)col * N_DIM + row] = v;
        s += v; sa += fabsf(v);
    }
}

// ---------------------------------------------------------------------------
// corr = Zt Zt^T, upper-triangular 32x32 groups, 2x2 WMMA tiles per wave,
// software-pipelined K-loop (8 loads in flight behind 4 WMMAs).
// ---------------------------------------------------------------------------
__global__ void __launch_bounds__(256) corr_gemm_kernel(
    const _Float16* __restrict__ ZT, float* __restrict__ corrM,
    float* __restrict__ partials)
{
    const int wave = threadIdx.x >> 5, lane = threadIdx.x & 31;
    const int q = blockIdx.x * 8 + wave;
    int gi, gj; group_ij(q, gi, gj);
    const int m = lane & 15, hi = lane >> 4;

    const _Float16* a0p = ZT + (size_t)(gi * 32 + m) * T_DIM;
    const _Float16* a1p = a0p + (size_t)16 * T_DIM;
    const _Float16* b0p = ZT + (size_t)(gj * 32 + m) * T_DIM;
    const _Float16* b1p = b0p + (size_t)16 * T_DIM;

    v8f c00 = {}, c01 = {}, c10 = {}, c11 = {};
    v16h a0 = load_fragA(a0p, 0, hi), a1 = load_fragA(a1p, 0, hi);
    v16h b0 = load_fragB(b0p, 0, hi), b1 = load_fragB(b1p, 0, hi);

    for (int k = 0; k < T_DIM; k += 32) {
        const int kn = (k + 32) & (T_DIM - 1);      // wrap: last preload harmless
        __builtin_prefetch(a0p + k + 512, 0, 1);    // global_prefetch_b8
        __builtin_prefetch(a1p + k + 512, 0, 1);
        __builtin_prefetch(b0p + k + 512, 0, 1);
        __builtin_prefetch(b1p + k + 512, 0, 1);
        v16h na0 = load_fragA(a0p, kn, hi);
        v16h na1 = load_fragA(a1p, kn, hi);
        v16h nb0 = load_fragB(b0p, kn, hi);
        v16h nb1 = load_fragB(b1p, kn, hi);
        c00 = wmma_f16(a0, b0, c00);
        c01 = wmma_f16(a0, b1, c01);
        c10 = wmma_f16(a1, b0, c10);
        c11 = wmma_f16(a1, b1, c11);
        a0 = na0; a1 = na1; b0 = nb0; b1 = nb1;
    }

    float s = 0.f, sa = 0.f;
    const bool mir = (gi != gj);
    store_tile(corrM, c00, gi * 32,      gj * 32,      lane, mir, s, sa);
    store_tile(corrM, c01, gi * 32,      gj * 32 + 16, lane, mir, s, sa);
    store_tile(corrM, c10, gi * 32 + 16, gj * 32,      lane, mir, s, sa);
    store_tile(corrM, c11, gi * 32 + 16, gj * 32 + 16, lane, mir, s, sa);
    for (int off = 16; off; off >>= 1) {
        s  += __shfl_down(s,  off);
        sa += __shfl_down(sa, off);
    }
    if (lane == 0) {
        float f = mir ? 2.f : 1.f;
        partials[2 * q] = s * f;
        partials[2 * q + 1] = sa * f;
    }
}

// ---------------------------------------------------------------------------
// eq group sums: eq = 1.5*|S|^T|S| + 0.5*S^T S + T - n_i - n_j.
// 2x2 tiles, 8 WMMAs per K-step, pipelined. Only scalar group sums kept.
// ---------------------------------------------------------------------------
__global__ void __launch_bounds__(256) eq_gemm_kernel(
    const _Float16* __restrict__ ST, const float* __restrict__ nz,
    float* __restrict__ eq_partials)
{
    const int wave = threadIdx.x >> 5, lane = threadIdx.x & 31;
    const int q = blockIdx.x * 8 + wave;
    int gi, gj; group_ij(q, gi, gj);
    const int m = lane & 15, hi = lane >> 4;

    const _Float16* a0p = ST + (size_t)(gi * 32 + m) * T_DIM;
    const _Float16* a1p = a0p + (size_t)16 * T_DIM;
    const _Float16* b0p = ST + (size_t)(gj * 32 + m) * T_DIM;
    const _Float16* b1p = b0p + (size_t)16 * T_DIM;

    v8f p00 = {}, p01 = {}, p10 = {}, p11 = {};   // S^T S
    v8f q00 = {}, q01 = {}, q10 = {}, q11 = {};   // |S|^T |S|
    v16h a0 = load_fragA(a0p, 0, hi), a1 = load_fragA(a1p, 0, hi);
    v16h b0 = load_fragB(b0p, 0, hi), b1 = load_fragB(b1p, 0, hi);

    for (int k = 0; k < T_DIM; k += 32) {
        const int kn = (k + 32) & (T_DIM - 1);
        __builtin_prefetch(a0p + k + 512, 0, 1);
        __builtin_prefetch(a1p + k + 512, 0, 1);
        __builtin_prefetch(b0p + k + 512, 0, 1);
        __builtin_prefetch(b1p + k + 512, 0, 1);
        v16h na0 = load_fragA(a0p, kn, hi);
        v16h na1 = load_fragA(a1p, kn, hi);
        v16h nb0 = load_fragB(b0p, kn, hi);
        v16h nb1 = load_fragB(b1p, kn, hi);
        v16h aa0 = habs16(a0), aa1 = habs16(a1);
        v16h ba0 = habs16(b0), ba1 = habs16(b1);
        p00 = wmma_f16(a0, b0, p00);
        p01 = wmma_f16(a0, b1, p01);
        p10 = wmma_f16(a1, b0, p10);
        p11 = wmma_f16(a1, b1, p11);
        q00 = wmma_f16(aa0, ba0, q00);
        q01 = wmma_f16(aa0, ba1, q01);
        q10 = wmma_f16(aa1, ba0, q10);
        q11 = wmma_f16(aa1, ba1, q11);
        a0 = na0; a1 = na1; b0 = nb0; b1 = nb1;
    }

    float s1 = 0.f, s2 = 0.f;
    #pragma unroll
    for (int r = 0; r < 8; ++r) {
        s1 += p00[r] + p01[r] + p10[r] + p11[r];
        s2 += q00[r] + q01[r] + q10[r] + q11[r];
    }
    // 32 rows + 32 cols of nz, two per lane
    float nc = nz[gi * 32 + lane] + nz[gj * 32 + lane];
    for (int off = 16; off; off >>= 1) {
        s1 += __shfl_down(s1, off);
        s2 += __shfl_down(s2, off);
        nc += __shfl_down(nc, off);
    }
    if (lane == 0) {
        float grp = 1.5f * s2 + 0.5f * s1
                  + 1024.f * (float)T_DIM - 32.f * nc;
        eq_partials[q] = (gi != gj) ? 2.f * grp : grp;
    }
}

// ---------------------------------------------------------------------------
// Rolling off-diagonal mean of window corrcoef. One block per window start.
// Window length is a template param so per-t accumulators stay in registers.
// ---------------------------------------------------------------------------
template <int W>
__global__ void __launch_bounds__(256) rolling_kernel(
    const float* __restrict__ XT, float* __restrict__ out, int nwin)
{
    const int s = blockIdx.x;
    if (s >= nwin) return;
    const int tid = threadIdx.x, lane = tid & 31, wave = tid >> 5;
    __shared__ float red[256];
    __shared__ float part[W][8];
    __shared__ float rsq[W];
    __shared__ float Csh;

    float muinv = 0.f;
    float rloc[W];
    #pragma unroll
    for (int t = 0; t < W; ++t) rloc[t] = 0.f;

    #pragma unroll
    for (int cc = 0; cc < 4; ++cc) {
        const int c = tid * 4 + cc;
        const float* xp = XT + (size_t)c * T_DIM + s;
        float sum = 0.f, ssq = 0.f;
        #pragma unroll
        for (int t = 0; t < W; ++t) { float v = xp[t]; sum += v; ssq += v * v; }
        float mu = sum / (float)W;
        float iv = rsqrtf(fmaxf(ssq - sum * mu, 1e-30f));
        muinv += mu * iv;
        #pragma unroll
        for (int t = 0; t < W; ++t) rloc[t] += xp[t] * iv;
    }

    red[tid] = muinv; __syncthreads();
    for (int off = 128; off; off >>= 1) {
        if (tid < off) red[tid] += red[tid + off];
        __syncthreads();
    }
    if (tid == 0) Csh = red[0];

    #pragma unroll
    for (int t = 0; t < W; ++t) {
        float p = rloc[t];
        for (int off = 16; off; off >>= 1) p += __shfl_down(p, off);
        if (lane == 0) part[t][wave] = p;
    }
    __syncthreads();
    if (tid < W) {
        float r = part[tid][0] + part[tid][1] + part[tid][2] + part[tid][3]
                + part[tid][4] + part[tid][5] + part[tid][6] + part[tid][7] - Csh;
        rsq[tid] = r * r;
    }
    __syncthreads();
    if (tid == 0) {
        float tot = 0.f;
        #pragma unroll
        for (int t = 0; t < W; ++t) tot += rsq[t];
        out[s] = (tot - (float)N_DIM) / ((float)N_DIM * (float)(N_DIM - 1));
    }
}

// ---------------------------------------------------------------------------
// Grid-parallel power iteration: 32 blocks, deterministic fixed-order
// reductions, atomic-counter grid barrier (all blocks resident on MI455X).
// ---------------------------------------------------------------------------
__global__ void zero_ctr_kernel(unsigned* c) { *c = 0u; }

__device__ __forceinline__ void grid_barrier(unsigned* ctr, unsigned* round) {
    __syncthreads();
    if (threadIdx.x == 0) {
        __threadfence();
        atomicAdd(ctr, 1u);
        *round += 1u;
        const unsigned goal = (*round) * (unsigned)PI_BLOCKS;
        while (atomicAdd(ctr, 0u) < goal) { }
        __threadfence();
    }
    __syncthreads();
}

__global__ void __launch_bounds__(256) power_iter_kernel(
    const float* __restrict__ C, float* __restrict__ xbuf,
    float* __restrict__ ybuf, float* __restrict__ red_part,
    unsigned* __restrict__ ctr, float* __restrict__ eig_out)
{
    const int b = blockIdx.x, tid = threadIdx.x;
    const int seg = tid & 7;                 // 8 threads per row
    const int r = b * 32 + (tid >> 3);       // 32 rows per block
    __shared__ float lds[256];
    unsigned round = 0;

    if (tid < 32) xbuf[b * 32 + tid] = rsqrtf((float)N_DIM);
    grid_barrier(ctr, &round);

    float lam = 0.f;
    for (int it = 0; it < PI_ITERS; ++it) {
        const float* row = C + (size_t)r * N_DIM + seg * 128;
        const float* xv  = xbuf + seg * 128;
        float p = 0.f;
        for (int j = 0; j < 128; j += 4) {
            float4 cv = *(const float4*)(row + j);
            float4 xw = *(const float4*)(xv + j);
            p += cv.x * xw.x + cv.y * xw.y + cv.z * xw.z + cv.w * xw.w;
        }
        p += __shfl_down(p, 4, 8);
        p += __shfl_down(p, 2, 8);
        p += __shfl_down(p, 1, 8);
        float ss = 0.f;
        if (seg == 0) { ybuf[r] = p; ss = p * p; }
        lds[tid] = ss; __syncthreads();
        for (int off = 128; off; off >>= 1) {
            if (tid < off) lds[tid] += lds[tid + off];
            __syncthreads();
        }
        if (tid == 0) red_part[b] = lds[0];
        grid_barrier(ctr, &round);

        if (tid == 0) {
            float t = 0.f;
            #pragma unroll
            for (int i = 0; i < PI_BLOCKS; ++i) t += red_part[i];
            lds[0] = sqrtf(t);
        }
        __syncthreads();
        const float nrm = lds[0];
        lam = nrm;
        if (tid < 32) xbuf[b * 32 + tid] = ybuf[b * 32 + tid] / nrm;
        grid_barrier(ctr, &round);
    }
    if (b == 0 && tid == 0) eig_out[0] = lam;
}

// ---------------------------------------------------------------------------
// Finalize: all scalar features + tiny MLP. Single 256-thread block.
// ---------------------------------------------------------------------------
__device__ float fin_block_sum(float v) {
    __shared__ float red[256];
    const int tid = threadIdx.x;
    red[tid] = v; __syncthreads();
    for (int off = 128; off; off >>= 1) {
        if (tid < off) red[tid] += red[tid + off];
        __syncthreads();
    }
    float r = red[0]; __syncthreads();
    return r;
}
__device__ float fin_block_max(float v) {
    __shared__ float red[256];
    const int tid = threadIdx.x;
    red[tid] = v; __syncthreads();
    for (int off = 128; off; off >>= 1) {
        if (tid < off) red[tid] = fmaxf(red[tid], red[tid + off]);
        __syncthreads();
    }
    float r = red[0]; __syncthreads();
    return r;
}

__global__ void __launch_bounds__(256) finalize_kernel(
    const float* __restrict__ returns, const float* __restrict__ positions,
    const float* __restrict__ W1, const float* __restrict__ b1,
    const float* __restrict__ g,  const float* __restrict__ be,
    const float* __restrict__ bm, const float* __restrict__ bv,
    const float* __restrict__ W2, const float* __restrict__ b2,
    const float* __restrict__ W3, const float* __restrict__ b3,
    const float* __restrict__ cs_std, const float* __restrict__ corr_part,
    const float* __restrict__ eq_part, const float* __restrict__ roll20,
    const float* __restrict__ roll10, const float* __restrict__ eig,
    float* __restrict__ out)
{
    const int tid = threadIdx.x;
    const float Nf = (float)N_DIM;

    // herding index (diff-mean telescopes)
    float s = 0.f;
    for (int i = tid; i < T_DIM; i += 256) s += cs_std[i];
    float cs_mean = fin_block_sum(s) / (float)T_DIM;
    float trend = (cs_std[0] - cs_std[T_DIM - 1]) / (float)(T_DIM - 1);
    float herding_index = clampf(trend / (cs_mean + 1e-6f) + 0.5f, 0.f, 1.f);

    // corr reductions over triangular group partials (trace == N analytically)
    float ps = 0.f, pa = 0.f;
    for (int i = tid; i < NGROUPS; i += 256) { ps += corr_part[2 * i]; pa += corr_part[2 * i + 1]; }
    float corr_sum = fin_block_sum(ps);
    float corr_abs = fin_block_sum(pa);
    float avg_corr = (corr_sum - Nf) / (Nf * (Nf - 1.f));
    float return_div = 1.f - corr_abs / (Nf * Nf);

    // phase coupling (diag(eq) == T analytically)
    float es = 0.f;
    for (int i = tid; i < NGROUPS; i += 256) es += eq_part[i];
    float eq_tot = fin_block_sum(es);
    float avg_conc = (eq_tot / (float)T_DIM - Nf) / (Nf * (Nf - 1.f));
    float phase_coupling = clampf((avg_conc - 0.5f) * 2.f, 0.f, 1.f);

    // phase locking from roll20
    float cnt = 0.f;
    for (int i = tid; i < T_DIM - 20; i += 256) cnt += (roll20[i] > 0.7f) ? 1.f : 0.f;
    float phase_locking = fin_block_sum(cnt) / (float)(T_DIM - 20);

    // correlation surge from roll10
    const int NW10 = T_DIM - 10;
    float rs_rec = 0.f, rs_hist = 0.f;
    for (int i = tid; i < NW10; i += 256) {
        float v = roll10[i];
        if (i >= NW10 - 5) rs_rec += v; else rs_hist += v;
    }
    float recent = fin_block_sum(rs_rec) / 5.f;
    float hist   = fin_block_sum(rs_hist) / (float)(NW10 - 5);
    float surge  = (hist > 0.f) ? clampf((recent - hist) / hist, 0.f, 1.f) : 0.f;

    // position diversity
    float asum = 0.f, amax = 0.f;
    for (int i = tid; i < N_DIM; i += 256) {
        float v = fabsf(positions[i]);
        asum += v; amax = fmaxf(amax, v);
    }
    asum = fin_block_sum(asum);
    amax = fin_block_max(amax);
    float pos_div = 1.f - amax / asum;
    float diversity_loss = 1.f - sqrtf(fmaxf(return_div * pos_div, 0.f));

    float sync_risk = fminf(1.f, (eig[0] / Nf) * avg_corr);

    // MLP: [returns[-1], positions] -> 128 -> BN -> 64 -> 3 -> softmax[2]
    __shared__ float feat[2 * N_DIM];
    __shared__ float h1s[128];
    __shared__ float h2s[64];
    __shared__ float lg[3];
    for (int i = tid; i < N_DIM; i += 256) {
        feat[i]         = returns[(size_t)(T_DIM - 1) * N_DIM + i];
        feat[N_DIM + i] = positions[i];
    }
    __syncthreads();
    if (tid < 128) {
        float a = b1[tid];
        const float* w = W1 + (size_t)tid * (2 * N_DIM);
        for (int j = 0; j < 2 * N_DIM; ++j) a += w[j] * feat[j];
        a = fmaxf(a, 0.f);
        a = (a - bm[tid]) * rsqrtf(bv[tid] + 1e-5f) * g[tid] + be[tid];
        h1s[tid] = a;
    }
    __syncthreads();
    if (tid < 64) {
        float a = b2[tid];
        const float* w = W2 + (size_t)tid * 128;
        for (int j = 0; j < 128; ++j) a += w[j] * h1s[j];
        h2s[tid] = fmaxf(a, 0.f);
    }
    __syncthreads();
    if (tid < 3) {
        float a = b3[tid];
        const float* w = W3 + (size_t)tid * 64;
        for (int j = 0; j < 64; ++j) a += w[j] * h2s[j];
        lg[tid] = a;
    }
    __syncthreads();
    if (tid == 0) {
        float mx = fmaxf(lg[0], fmaxf(lg[1], lg[2]));
        float e0 = expf(lg[0] - mx), e1 = expf(lg[1] - mx), e2 = expf(lg[2] - mx);
        float severity = e2 / (e0 + e1 + e2);
        float collective = (herding_index + sync_risk + diversity_loss) / 3.f;
        out[0] = herding_index;
        out[1] = severity;
        out[2] = sync_risk;
        out[3] = phase_locking;
        out[4] = diversity_loss;
        out[5] = surge;
        out[6] = phase_coupling;
        out[7] = collective;
    }
}

// ---------------------------------------------------------------------------
// Host launcher
// ---------------------------------------------------------------------------
extern "C" void kernel_launch(void* const* d_in, const int* in_sizes, int n_in,
                              void* d_out, int out_size, void* d_ws, size_t ws_size,
                              hipStream_t stream)
{
    (void)in_sizes; (void)n_in; (void)out_size; (void)ws_size;

    const float* X   = (const float*)d_in[0];   // [T, N]
    const float* pos = (const float*)d_in[1];   // [N]
    const float* W1  = (const float*)d_in[2];
    const float* b1  = (const float*)d_in[3];
    const float* g   = (const float*)d_in[4];
    const float* be  = (const float*)d_in[5];
    const float* bm  = (const float*)d_in[6];
    const float* bv  = (const float*)d_in[7];
    const float* W2  = (const float*)d_in[8];
    const float* b2  = (const float*)d_in[9];
    const float* W3  = (const float*)d_in[10];
    const float* b3  = (const float*)d_in[11];
    float* out = (float*)d_out;

    char* ws = (char*)d_ws;
    size_t off = 0;
    auto alloc = [&](size_t bytes) -> void* {
        off = (off + 255) & ~(size_t)255;
        void* p = ws + off;
        off += bytes;
        return p;
    };

    _Float16* ZT     = (_Float16*)alloc((size_t)N_DIM * T_DIM * sizeof(_Float16));
    _Float16* STm    = (_Float16*)alloc((size_t)N_DIM * T_DIM * sizeof(_Float16));
    float* corrM     = (float*)alloc((size_t)N_DIM * N_DIM * sizeof(float));
    float* XT        = (float*)alloc((size_t)N_DIM * T_DIM * sizeof(float));
    float* colmean   = (float*)alloc(N_DIM * sizeof(float));
    float* colinv    = (float*)alloc(N_DIM * sizeof(float));
    float* nz        = (float*)alloc(N_DIM * sizeof(float));
    float* cs_std    = (float*)alloc(T_DIM * sizeof(float));
    float* corr_part = (float*)alloc(2 * NGROUPS * sizeof(float));
    float* eq_part   = (float*)alloc(NGROUPS * sizeof(float));
    float* roll20    = (float*)alloc(T_DIM * sizeof(float));
    float* roll10    = (float*)alloc(T_DIM * sizeof(float));
    float* xbuf      = (float*)alloc(N_DIM * sizeof(float));
    float* ybuf      = (float*)alloc(N_DIM * sizeof(float));
    float* red_part  = (float*)alloc(PI_BLOCKS * sizeof(float));
    unsigned* ctr    = (unsigned*)alloc(256);
    float* eig       = (float*)alloc(256);

    transpose_kernel<<<dim3(N_DIM / 32, T_DIM / 32), dim3(32, 8), 0, stream>>>(X, XT);
    col_stats_kernel<<<N_DIM, 256, 0, stream>>>(XT, colmean, colinv, nz);
    encode_kernel<<<(N_DIM * T_DIM) / (256 * 8), 256, 0, stream>>>(
        XT, colmean, colinv, ZT, STm);
    row_std_kernel<<<T_DIM / 8, 256, 0, stream>>>(X, cs_std);
    corr_gemm_kernel<<<NGROUPS / 8, 256, 0, stream>>>(ZT, corrM, corr_part);
    eq_gemm_kernel<<<NGROUPS / 8, 256, 0, stream>>>(STm, nz, eq_part);
    rolling_kernel<20><<<T_DIM - 20, 256, 0, stream>>>(XT, roll20, T_DIM - 20);
    rolling_kernel<10><<<T_DIM - 10, 256, 0, stream>>>(XT, roll10, T_DIM - 10);
    zero_ctr_kernel<<<1, 1, 0, stream>>>(ctr);
    power_iter_kernel<<<PI_BLOCKS, 256, 0, stream>>>(
        corrM, xbuf, ybuf, red_part, ctr, eig);
    finalize_kernel<<<1, 256, 0, stream>>>(
        X, pos, W1, b1, g, be, bm, bv, W2, b2, W3, b3,
        cs_std, corr_part, eq_part, roll20, roll10, eig, out);
}
